// BPTAttentionWrapperNeoX_3487513444834
// MI455X (gfx1250) — matrix operand order repaired and checked by reading, not compiled, not run on hardware
//
#include <hip/hip_runtime.h>
#include <hip/hip_bf16.h>

// ---------------------------------------------------------------------------
// NeoX attention block for MI455X (gfx1250, wave32, WMMA, async-to-LDS DMA).
//
// Pipeline:
//   0) one-shot f32->bf16 conversion of X, W_qkv, W_dense (pure streaming,
//      halves L2 operand footprint) + RoPE cos/sin table (S x 16).
//   1) QKV GEMM: double-buffered GLOBAL_LOAD_ASYNC_TO_LDS_B128 staging,
//      v_wmma_f32_16x16x32_bf16 compute, fused bias+RoPE epilogue scattering
//      q/k/v bf16 into (B,H,S,HS).
//   2) flash attention per (b,h): Q frags in registers, K tile transposed via
//      LDS, online softmax, P relayout through LDS, V B-frags from global.
//   3) dense GEMM (same async pipeline), bias, f32 output.
// ---------------------------------------------------------------------------

#define HEADS   16
#define HS      128
#define SEQ     2048
#define BATCH   2
#define DMODEL  2048
#define NQKV    6144   // 3 * HEADS * HS

typedef __attribute__((ext_vector_type(16))) __bf16 v16bf;
typedef __attribute__((ext_vector_type(8)))  float  v8f;

union Frag16 { v16bf v; unsigned int u[8]; };
static_assert(sizeof(Frag16) == 32, "frag size");

__device__ __forceinline__ unsigned short f2bf(float f) {
  union { float f; unsigned int u; } x; x.f = f;
  unsigned int r = 0x7FFFu + ((x.u >> 16) & 1u);   // round-to-nearest-even
  return (unsigned short)((x.u + r) >> 16);
}

__device__ __forceinline__ v8f wmma_bf16(const Frag16& a, const Frag16& b, v8f c) {
  return __builtin_amdgcn_wmma_f32_16x16x32_bf16(false, a.v, false, b.v,
                                                 (short)0, c, false, false);
}

// generic pointer (into LDS) -> raw 32-bit LDS offset for async-DMA operands
typedef const __attribute__((address_space(3))) void* lds_cptr_t;
__device__ __forceinline__ unsigned lds_addr(const void* p) {
  return (unsigned)(unsigned long long)(lds_cptr_t)p;
}

// one async b128 copy: 16 bytes/lane, global -> LDS, tracked by ASYNCcnt
__device__ __forceinline__ void async_b128(unsigned lds_off, const void* gptr) {
  asm volatile("global_load_async_to_lds_b128 %0, %1, off"
               :: "v"(lds_off), "v"(gptr) : "memory");
}

// ---------------------------------------------------------------------------
// Kernel 0a: f32 -> bf16 bulk conversion (vectorized 4-wide)
// ---------------------------------------------------------------------------
__global__ __launch_bounds__(256) void cvt_bf16_kernel(
    const float* __restrict__ src, unsigned short* __restrict__ dst, int n4)
{
  int i = blockIdx.x * blockDim.x + threadIdx.x;
  if (i < n4) {
    float4 f = ((const float4*)src)[i];
    uint2 o;
    o.x = (unsigned)f2bf(f.x) | ((unsigned)f2bf(f.y) << 16);
    o.y = (unsigned)f2bf(f.z) | ((unsigned)f2bf(f.w) << 16);
    ((uint2*)dst)[i] = o;
  }
}

// ---------------------------------------------------------------------------
// Kernel 0b: RoPE cache, cos/sin[s][j], j = 0..15, inv_freq = 1e4^(-j/16)
// ---------------------------------------------------------------------------
__global__ __launch_bounds__(256) void rope_tab_kernel(
    float* __restrict__ cosT, float* __restrict__ sinT)
{
  int i = blockIdx.x * blockDim.x + threadIdx.x;   // SEQ*16 threads
  int s = i >> 4, j = i & 15;
  float ang = (float)s * expf(-(float)j * 0.57564627324851142f); // ln(1e4)/16
  cosT[i] = cosf(ang);
  sinT[i] = sinf(ang);
}

// ---------------------------------------------------------------------------
// Shared GEMM core: C(64x64) += A(64xK) * B(Kx64), bf16 operands, K=DMODEL.
// Double-buffered async-to-LDS staging. 4 waves, wave w owns rows 16w..16w+15.
// LDS pitches: 80B / 144B rows -> 16B aligned for B128 DMA, odd-ish dword
// strides for mostly conflict-free fragment reads.
// ---------------------------------------------------------------------------
#define APITCH 40   // ushorts per A-tile row (64 rows x 32 k)
#define BPITCH 72   // ushorts per B-tile row (32 k-rows x 64 n)

__device__ __forceinline__ void gemm_tiles_async(
    const unsigned short* __restrict__ Ag,   // M x DMODEL row-major bf16
    const unsigned short* __restrict__ Bg,   // DMODEL x N row-major bf16
    int N, int mBase, int nBase,
    unsigned short* AsBase, unsigned short* BsBase,   // 2 LDS buffers each
    v8f acc[4])
{
  const int tid  = threadIdx.x;
  const int lane = tid & 31;
  const int wave = tid >> 5;
  const int half = lane >> 4;
  const int nl   = lane & 15;

  auto issue = [&](int buf, int k0g) {
    unsigned short* As = AsBase + buf * (64 * APITCH);
    unsigned short* Bs = BsBase + buf * (32 * BPITCH);
    #pragma unroll
    for (int j = 0; j < 2; ++j) {          // A tile: 256 x 16B chunks
      int c = tid + j * 128;
      int r = c >> 2, seg = c & 3;
      async_b128(lds_addr(As + r * APITCH + seg * 8),
                 Ag + (size_t)(mBase + r) * DMODEL + k0g + seg * 8);
    }
    #pragma unroll
    for (int j = 0; j < 2; ++j) {          // B tile: 256 x 16B chunks
      int c = tid + j * 128;
      int kl = c >> 3, seg = c & 7;
      async_b128(lds_addr(Bs + kl * BPITCH + seg * 8),
                 Bg + (size_t)(k0g + kl) * N + nBase + seg * 8);
    }
  };

  issue(0, 0);
  for (int it = 0; it < DMODEL / 32; ++it) {
    int cur = it & 1;
    if (it + 1 < DMODEL / 32) {
      issue(cur ^ 1, (it + 1) * 32);                       // prefetch next tile
      asm volatile("s_wait_asynccnt 0x4" ::: "memory");    // batch `it` done
    } else {
      asm volatile("s_wait_asynccnt 0x0" ::: "memory");
    }
    __syncthreads();

    const unsigned short* As = AsBase + cur * (64 * APITCH);
    const unsigned short* Bs = BsBase + cur * (32 * BPITCH);

    Frag16 a;   // A frag: lane = M row, K pattern per ISA 16-bit A layout
    #pragma unroll
    for (int v = 0; v < 8; ++v) {
      int kk = ((v < 4) ? 0 : 16) + (half << 3) + ((v & 3) << 1);
      a.u[v] = *(const unsigned int*)(As + (wave * 16 + nl) * APITCH + kk);
    }
    #pragma unroll
    for (int t = 0; t < 4; ++t) {
      Frag16 b; // B frag: lane = N col, K pattern same as A frag
      #pragma unroll
      for (int v = 0; v < 8; ++v) {
        int kk = ((v < 4) ? 0 : 16) + (half << 3) + ((v & 3) << 1);
        b.u[v] = (unsigned)Bs[kk * BPITCH + t * 16 + nl] | ((unsigned)Bs[(kk + 1) * BPITCH + t * 16 + nl] << 16);
      }
      acc[t] = wmma_bf16(a, b, acc[t]);
    }
    __syncthreads();
  }
}

// ---------------------------------------------------------------------------
// Kernel 1: QKV GEMM + bias + RoPE, scatter q/k/v bf16 into (B,H,S,HS)
// ---------------------------------------------------------------------------
__global__ __launch_bounds__(128) void qkv_rope_kernel(
    const unsigned short* __restrict__ Xb, const int* __restrict__ pos,
    const unsigned short* __restrict__ Wb, const float* __restrict__ bias,
    const float* __restrict__ cosT, const float* __restrict__ sinT,
    unsigned short* __restrict__ qb, unsigned short* __restrict__ kb,
    unsigned short* __restrict__ vb)
{
  __shared__ unsigned short As[2][64 * APITCH];
  __shared__ unsigned short Bs[2][32 * BPITCH];

  const int tid  = threadIdx.x;
  const int lane = tid & 31;
  const int wave = tid >> 5;
  const int half = lane >> 4;
  const int nl   = lane & 15;

  const int nBase = blockIdx.x * 64;
  const int mBase = blockIdx.y * 64;

  v8f acc[4] = {};
  gemm_tiles_async(Xb, Wb, NQKV, mBase, nBase, &As[0][0], &Bs[0][0], acc);

  // bias, then RoPE on first 32 dims of q,k; pairs (d, d+-16) sit in the
  // adjacent n-tiles t and t^1 of this same wave (head stride 384 = 6*64).
  float val[4][8];
  #pragma unroll
  for (int t = 0; t < 4; ++t) {
    float bv = bias[nBase + t * 16 + nl];
    #pragma unroll
    for (int r = 0; r < 8; ++r) val[t][r] = acc[t][r] + bv;
  }

  #pragma unroll
  for (int t = 0; t < 4; ++t) {
    int col = nBase + t * 16 + nl;
    int h = col / 384;
    int rem = col - h * 384;
    int which = rem >> 7;       // 0=q 1=k 2=v
    int d = rem & 127;
    #pragma unroll
    for (int r = 0; r < 8; ++r) {
      int grow = mBase + wave * 16 + r + (half << 3);   // b*SEQ + s
      int b = grow >> 11;
      int s = grow & (SEQ - 1);
      float vv = val[t][r];
      if (which < 2 && d < 32) {
        float partner = val[t ^ 1][r];
        int ti = pos[grow] * 16 + (d & 15);
        float cs = cosT[ti], sn = sinT[ti];
        vv = (d < 16) ? (vv * cs - partner * sn)    // rotate_half: -x2
                      : (vv * cs + partner * sn);   //               x1
      }
      unsigned short o = f2bf(vv);
      size_t idx = ((size_t)(b * HEADS + h) * SEQ + s) * HS + d;
      if (which == 0)      qb[idx] = o;
      else if (which == 1) kb[idx] = o;
      else                 vb[idx] = o;
    }
  }
}

// ---------------------------------------------------------------------------
// Kernel 2: flash attention per (b,h). Block = 64 q-rows (4 waves x 16).
// ---------------------------------------------------------------------------
#define KSP 34  // Ks pitch: 128 d-rows x 32 k  (68B rows: 17-dword stride)
#define PSP 34  // Ps pitch: per-wave 16 x 32

__global__ __launch_bounds__(128) void attn_kernel(
    const unsigned short* __restrict__ qb, const unsigned short* __restrict__ kb,
    const unsigned short* __restrict__ vb, unsigned short* __restrict__ attn)
{
  __shared__ unsigned short Ks[HS * KSP];
  __shared__ unsigned short Ps[4 * 16 * PSP];

  const int tid  = threadIdx.x;
  const int lane = tid & 31;
  const int wave = tid >> 5;
  const int half = lane >> 4;
  const int nl   = lane & 15;

  const int bh = blockIdx.y;
  const int b  = bh >> 4;
  const int h  = bh & 15;

  const unsigned short* qp = qb + (size_t)bh * SEQ * HS;
  const unsigned short* kp = kb + (size_t)bh * SEQ * HS;
  const unsigned short* vp = vb + (size_t)bh * SEQ * HS;

  const int qBase = blockIdx.x * 64 + wave * 16;

  // Q A-fragments, resident for the whole kernel (16 rows x 128 d)
  Frag16 aq[4];
  {
    int qrow = qBase + nl;
    #pragma unroll
    for (int f = 0; f < 4; ++f)
      #pragma unroll
      for (int v = 0; v < 8; ++v) {
        int d0 = f * 32 + ((v < 4) ? 0 : 16) + (half << 3) + ((v & 3) << 1);
        aq[f].u[v] = *(const unsigned int*)(qp + (size_t)qrow * HS + d0);
      }
  }

  v8f o[8] = {};
  float mrow[8], lrow[8];
  #pragma unroll
  for (int r = 0; r < 8; ++r) { mrow[r] = -1e30f; lrow[r] = 0.f; }

  for (int kb32 = 0; kb32 < SEQ; kb32 += 32) {
    if (kb32 + 32 < SEQ) {   // hint next K/V block toward the caches
      __builtin_prefetch(kp + (size_t)(kb32 + 32 + lane) * HS, 0, 1);
      __builtin_prefetch(vp + (size_t)(kb32 + 32 + lane) * HS, 0, 1);
    }
    __syncthreads();
    { // stage K tile transposed: Ks[d][k_local]
      int sl = tid & 31, dB = (tid >> 5) * 32;
      const unsigned short* src = kp + (size_t)(kb32 + sl) * HS + dB;
      #pragma unroll
      for (int j = 0; j < 32; ++j) Ks[(dB + j) * KSP + sl] = src[j];
    }
    __syncthreads();

    // S = Q * K^T : two 16x16 score tiles (cols kb32+16t .. +15)
    v8f sc[2];
    #pragma unroll
    for (int t = 0; t < 2; ++t) {
      v8f s = {};
      #pragma unroll
      for (int f = 0; f < 4; ++f) {
        Frag16 kf;
        #pragma unroll
        for (int v = 0; v < 8; ++v) {
          int kk = f * 32 + ((v < 4) ? 0 : 16) + (half << 3) + ((v & 3) << 1);
          kf.u[v] = (unsigned)Ks[kk * KSP + t * 16 + nl] | ((unsigned)Ks[(kk + 1) * KSP + t * 16 + nl] << 16);
        }
        s = wmma_bf16(aq[f], kf, s);
      }
      sc[t] = s;
    }

    const float scale = 0.08838834764831845f;   // 1/sqrt(128)
    #pragma unroll
    for (int r = 0; r < 8; ++r) {
      float s0 = sc[0][r] * scale;
      float s1 = sc[1][r] * scale;
      float mx = fmaxf(s0, s1);
      #pragma unroll
      for (int off = 1; off < 16; off <<= 1)      // row lives in a 16-lane group
        mx = fmaxf(mx, __shfl_xor(mx, off, 32));
      float mnew = fmaxf(mrow[r], mx);
      float corr = __expf(mrow[r] - mnew);
      float p0 = __expf(s0 - mnew);
      float p1 = __expf(s1 - mnew);
      float rs = p0 + p1;
      #pragma unroll
      for (int off = 1; off < 16; off <<= 1)
        rs += __shfl_xor(rs, off, 32);
      lrow[r] = lrow[r] * corr + rs;
      mrow[r] = mnew;
      #pragma unroll
      for (int dt = 0; dt < 8; ++dt) o[dt][r] *= corr;
      int m = r + (half << 3);
      Ps[(wave * 16 + m) * PSP + nl]      = f2bf(p0);   // C-layout -> LDS
      Ps[(wave * 16 + m) * PSP + 16 + nl] = f2bf(p1);
    }

    // relayout P into an A fragment (16x32)
    Frag16 ap;
    #pragma unroll
    for (int v = 0; v < 8; ++v) {
      int kk = ((v < 4) ? 0 : 16) + (half << 3) + ((v & 3) << 1);
      ap.u[v] = *(const unsigned int*)&Ps[(wave * 16 + nl) * PSP + kk];
    }

    // O += P * V  (B frag: lane = d column, K pattern same as A frag)
    #pragma unroll
    for (int dt = 0; dt < 8; ++dt) {
      Frag16 bv;
      const unsigned short* vr = vp + (size_t)kb32 * HS + dt * 16 + nl;
      #pragma unroll
      for (int v = 0; v < 8; ++v) {
        int kk = ((v < 4) ? 0 : 16) + (half << 3) + ((v & 3) << 1);
        bv.u[v] = (unsigned)vr[(size_t)kk * HS] | ((unsigned)vr[(size_t)(kk + 1) * HS] << 16);
      }
      o[dt] = wmma_bf16(ap, bv, o[dt]);
    }
  }

  // normalize and store into (B*S, H*HS) bf16 activation for the dense GEMM
  #pragma unroll
  for (int dt = 0; dt < 8; ++dt)
    #pragma unroll
    for (int r = 0; r < 8; ++r) {
      int qrow = qBase + r + (half << 3);
      float vv = o[dt][r] / lrow[r];
      attn[((size_t)(b * SEQ + qrow)) * (HEADS * HS) + h * HS + dt * 16 + nl] = f2bf(vv);
    }
}

// ---------------------------------------------------------------------------
// Kernel 3: out = attn @ W_dense + b_dense (bf16 operands, f32 output)
// ---------------------------------------------------------------------------
__global__ __launch_bounds__(128) void dense_kernel(
    const unsigned short* __restrict__ A, const unsigned short* __restrict__ Wb,
    const float* __restrict__ bias, float* __restrict__ out)
{
  __shared__ unsigned short As[2][64 * APITCH];
  __shared__ unsigned short Bs[2][32 * BPITCH];

  const int tid  = threadIdx.x;
  const int lane = tid & 31;
  const int wave = tid >> 5;
  const int half = lane >> 4;
  const int nl   = lane & 15;

  const int nBase = blockIdx.x * 64;
  const int mBase = blockIdx.y * 64;

  v8f acc[4] = {};
  gemm_tiles_async(A, Wb, DMODEL, mBase, nBase, &As[0][0], &Bs[0][0], acc);

  #pragma unroll
  for (int t = 0; t < 4; ++t) {
    int col = nBase + t * 16 + nl;
    float bv = bias[col];
    #pragma unroll
    for (int r = 0; r < 8; ++r) {
      int row = mBase + wave * 16 + r + (half << 3);
      out[(size_t)row * DMODEL + col] = acc[t][r] + bv;
    }
  }
}

// ---------------------------------------------------------------------------
extern "C" void kernel_launch(void* const* d_in, const int* in_sizes, int n_in,
                              void* d_out, int out_size, void* d_ws, size_t ws_size,
                              hipStream_t stream) {
  const float* X    = (const float*)d_in[0];   // (2,2048,2048) f32
  const int*   pos  = (const int*)  d_in[1];   // (2,2048) i32
  const float* Wqkv = (const float*)d_in[2];   // (2048, 6144) f32
  const float* bqkv = (const float*)d_in[3];   // (6144,) f32
  const float* Wd   = (const float*)d_in[4];   // (2048, 2048) f32
  const float* bd   = (const float*)d_in[5];   // (2048,) f32
  float* out = (float*)d_out;                  // (2,2048,2048) f32

  // workspace layout (ushort units). q/k/v/attn 64MiB + bf16 copies + tables.
  unsigned short* ws = (unsigned short*)d_ws;
  const size_t per   = (size_t)BATCH * HEADS * SEQ * HS;   // 8,388,608
  unsigned short* qb    = ws;                 // per
  unsigned short* kbuf  = ws + per;           // per
  unsigned short* vbuf  = ws + 2 * per;       // per
  unsigned short* attn  = ws + 3 * per;       // per
  unsigned short* Xb    = ws + 4 * per;       // 4096*2048      =  8,388,608
  unsigned short* Wqkvb = ws + 5 * per;       // 2048*6144      = 12,582,912
  unsigned short* Wdb   = Wqkvb + (size_t)DMODEL * NQKV;   // 2048*2048
  float* cosT = (float*)(Wdb + (size_t)DMODEL * DMODEL);   // SEQ*16 f32
  float* sinT = cosT + SEQ * 16;

  const int nX  = (BATCH * SEQ * DMODEL) / 4;   // float4 groups
  const int nWq = (DMODEL * NQKV) / 4;
  const int nWd = (DMODEL * DMODEL) / 4;
  cvt_bf16_kernel<<<(nX  + 255) / 256, 256, 0, stream>>>(X,    Xb,    nX);
  cvt_bf16_kernel<<<(nWq + 255) / 256, 256, 0, stream>>>(Wqkv, Wqkvb, nWq);
  cvt_bf16_kernel<<<(nWd + 255) / 256, 256, 0, stream>>>(Wd,   Wdb,   nWd);
  rope_tab_kernel<<<(SEQ * 16) / 256, 256, 0, stream>>>(cosT, sinT);

  qkv_rope_kernel<<<dim3(NQKV / 64, (BATCH * SEQ) / 64), 128, 0, stream>>>(
      Xb, pos, Wqkvb, bqkv, cosT, sinT, qb, kbuf, vbuf);
  attn_kernel<<<dim3(SEQ / 64, BATCH * HEADS), 128, 0, stream>>>(
      qb, kbuf, vbuf, attn);
  dense_kernel<<<dim3(DMODEL / 64, (BATCH * SEQ) / 64), 128, 0, stream>>>(
      attn, Wdb, bd, out);
}
